// OACNNs_2508260901481
// MI455X (gfx1250) — matrix operand order repaired and checked by reading, not compiled, not run on hardware
//
#include <hip/hip_runtime.h>
#include <cstdint>

// ---------------------------------------------------------------------------
// OACNN block for MI455X (gfx1250, wave32).  All N x 64 x 64 GEMMs (attention
// projections + 2x 27-tap submanifold convs) run on v_wmma_f32_16x16x32_bf16.
// Weight panels are stored transposed (n-major) in bf16 so B-fragments are
// contiguous b128 LDS reads; conv features are cached in bf16 so the 27-way
// gather is pure global_load_b128 traffic (L2-resident).  Weight tiles are
// staged into LDS with GLOBAL_LOAD_ASYNC_TO_LDS_B128 (ASYNCcnt) when the
// toolchain exposes the builtin, double-buffered against the WMMA loop.
// ---------------------------------------------------------------------------

typedef __bf16 bf16;
typedef __attribute__((ext_vector_type(16))) __bf16 v16bf;
typedef __attribute__((ext_vector_type(8)))  __bf16 v8bf;
typedef __attribute__((ext_vector_type(8)))  float  v8f;

#define C_DIM 64
#define KTAP  27

#if defined(__has_builtin)
#if __has_builtin(__builtin_amdgcn_global_load_async_to_lds_b128)
#define HAVE_ASYNC_LDS 1
#endif
#endif

__device__ __forceinline__ bf16 f2bf(float f) {
    unsigned u = __builtin_bit_cast(unsigned, f);
    unsigned r = (u + 0x7FFFu + ((u >> 16) & 1u)) >> 16;
    unsigned short s = (unsigned short)r;
    return __builtin_bit_cast(bf16, s);
}

__device__ __forceinline__ v8f wmma_bf16(v16bf a, v16bf b, v8f c) {
    // (neg_a, A, neg_b, B, c_mod, C, reuse_a, reuse_b)
    return __builtin_amdgcn_wmma_f32_16x16x32_bf16(false, a, false, b, (short)0, c, false, false);
}

__device__ __forceinline__ v16bf cat8(v8bf lo, v8bf hi) {
    return __builtin_shufflevector(lo, hi, 0, 1, 2, 3, 4, 5, 6, 7,
                                           8, 9, 10, 11, 12, 13, 14, 15);
}

// ---------------------------------------------------------------------------
// Weight-tile staging: global (transposed bf16 panel) -> LDS.
// NCHUNK = 16-byte chunks per thread (256 threads/block).
// Async path: one global_load_async_to_lds_b128 per chunk, ASYNCcnt-tracked.
// ---------------------------------------------------------------------------
template <int NCHUNK>
__device__ __forceinline__ void stage_tile(const bf16* __restrict__ src, bf16* dst, int tid) {
#ifdef HAVE_ASYNC_LDS
    typedef int g4i __attribute__((vector_size(16)));
    typedef __attribute__((address_space(1))) g4i* gp4;   // global side (AS1)
    typedef __attribute__((address_space(3))) g4i* lp4;   // LDS side (AS3)
    #pragma unroll
    for (int i = 0; i < NCHUNK; ++i) {
        bf16* g = const_cast<bf16*>(src) + (size_t)(tid + 256 * i) * 8;
        bf16* l = dst + (size_t)(tid + 256 * i) * 8;
        __builtin_amdgcn_global_load_async_to_lds_b128((gp4)g, (lp4)l, 0, 0);
    }
#else
    const unsigned* s = (const unsigned*)src;
    unsigned* d = (unsigned*)dst;
    #pragma unroll
    for (int i = 0; i < NCHUNK * 4; ++i) d[tid + 256 * i] = s[tid + 256 * i];
#endif
}

__device__ __forceinline__ void stage_wait() {
#ifdef HAVE_ASYNC_LDS
#if __has_builtin(__builtin_amdgcn_s_wait_asynccnt)
    __builtin_amdgcn_s_wait_asynccnt(0);
#else
    asm volatile("s_wait_asynccnt 0x0" ::: "memory");
#endif
#endif
    __syncthreads();
}

// orderable-uint encoding for float atomicMax
__device__ __forceinline__ unsigned fenc(float f) {
    unsigned u = __builtin_bit_cast(unsigned, f);
    return (u & 0x80000000u) ? ~u : (u | 0x80000000u);
}
__device__ __forceinline__ float fdec(unsigned u) {
    unsigned v = (u & 0x80000000u) ? (u & 0x7FFFFFFFu) : ~u;
    return __builtin_bit_cast(float, v);
}

// ---------------------------------------------------------------------------
// Conversions
// ---------------------------------------------------------------------------
__global__ void cvt_bf16_kernel(const float* __restrict__ src, bf16* __restrict__ dst, long count) {
    long i = (long)blockIdx.x * blockDim.x + threadIdx.x;
    if (i < count) dst[i] = f2bf(src[i]);
}

// transpose-convert weight panels: dst[mat][n][k] = src[mat][k][n]  (k-dim = kdim, n-dim = 64)
__global__ void cvt_bf16_T_kernel(const float* __restrict__ src, bf16* __restrict__ dst,
                                  int kdim, long count) {
    long i = (long)blockIdx.x * blockDim.x + threadIdx.x;
    if (i >= count) return;
    long per = (long)kdim * C_DIM;
    long mat = i / per;
    long rem = i - mat * per;
    long n   = rem / kdim;
    long k   = rem - n * kdim;
    dst[i] = f2bf(src[mat * per + k * C_DIM + n]);
}

// ---------------------------------------------------------------------------
// WMMA GEMM:  Y[N x 64] = X[N x KDIM] @ W[KDIM x 64]   (KDIM = 64 or 128)
// WbT is the transposed (n-major) bf16 weight panel.  For KDIM=128 the K axis
// is the concat [X0 | X1] (fuse layer).  8 waves/block, 16 rows/wave.
// ---------------------------------------------------------------------------
template <int KDIM>
__global__ __launch_bounds__(256)
void gemm_wmma_kernel(const float* __restrict__ X0, const float* __restrict__ X1,
                      const bf16* __restrict__ WbT, float* __restrict__ Y, long nrows) {
    __shared__ bf16 Wlds[KDIM * C_DIM];    // [n][k], n-major
    stage_tile<KDIM / 32>(WbT, Wlds, threadIdx.x);
    stage_wait();

    const int wave = threadIdx.x >> 5;
    const int lane = threadIdx.x & 31;
    const int hl   = lane >> 4;
    const int lr   = lane & 15;
    const long m0  = (long)blockIdx.x * 128 + wave * 16;
    long ma = m0 + lr;
    if (ma >= nrows) ma = nrows - 1;       // clamp; stores are guarded
    const float* xr0 = X0 + ma * C_DIM;
    const float* xr1 = X1 + ma * C_DIM;

    v8f acc[4] = {};
    #pragma unroll
    for (int kk = 0; kk < KDIM; kk += 32) {
        // A fragment: two contiguous 8-float chunks at kk+8*hl and kk+16+8*hl
        v16bf a;
        #pragma unroll
        for (int ch = 0; ch < 2; ++ch) {
            int cb = kk + 16 * ch + 8 * hl;
            const float* p = (cb < C_DIM) ? (xr0 + cb) : (xr1 + cb - C_DIM);
            float4 f0 = ((const float4*)p)[0];
            float4 f1 = ((const float4*)p)[1];
            int b0 = 8 * ch;
            a[b0 + 0] = f2bf(f0.x); a[b0 + 1] = f2bf(f0.y);
            a[b0 + 2] = f2bf(f0.z); a[b0 + 3] = f2bf(f0.w);
            a[b0 + 4] = f2bf(f1.x); a[b0 + 5] = f2bf(f1.y);
            a[b0 + 6] = f2bf(f1.z); a[b0 + 7] = f2bf(f1.w);
        }
        #pragma unroll
        for (int nt = 0; nt < 4; ++nt) {
            const v8bf* bp = (const v8bf*)&Wlds[(nt * 16 + lr) * KDIM + kk + 8 * hl];
            acc[nt] = wmma_bf16(a, cat8(bp[0], bp[2]), acc[nt]);
        }
    }
    #pragma unroll
    for (int nt = 0; nt < 4; ++nt) {
        #pragma unroll
        for (int r = 0; r < 8; ++r) {
            long row = m0 + r + 8 * hl;
            if (row < nrows) Y[row * C_DIM + nt * 16 + lr] = acc[nt][r];
        }
    }
}

// ---------------------------------------------------------------------------
// Submanifold conv:  H[n] = sum_k Fb[nbr[n,k]] @ Wc[k]      (gather-GEMM)
// Fb is the bf16 feature cache (gathers are pure b128 loads, L2-resident).
// WcT: 27 transposed 64x64 bf16 tiles, async double-buffered through LDS.
// 8 waves/block, 32 rows/wave (two A fragments share every B fragment).
// ---------------------------------------------------------------------------
__global__ __launch_bounds__(256)
void subm_conv_wmma_kernel(const bf16* __restrict__ Fb, const int* __restrict__ nbr,
                           const bf16* __restrict__ WcT, float* __restrict__ Y, long nrows) {
    __shared__ bf16 Wlds[2][C_DIM * C_DIM];
    const int tid  = threadIdx.x;
    const int wave = tid >> 5;
    const int lane = tid & 31;
    const int hl   = lane >> 4;
    const int lr   = lane & 15;
    const long m0  = (long)blockIdx.x * 256 + wave * 32;
    long r0 = m0 + lr;        if (r0 >= nrows) r0 = nrows - 1;
    long r1 = m0 + 16 + lr;   if (r1 >= nrows) r1 = nrows - 1;

    stage_tile<2>(WcT, Wlds[0], tid);     // preload tap 0
    stage_wait();

    v8f acc[2][4] = {};
    for (int tap = 0; tap < KTAP; ++tap) {
        const int cur = tap & 1;
        if (tap + 1 < KTAP)               // stage next tile into the other buffer
            stage_tile<2>(WcT + (size_t)(tap + 1) * C_DIM * C_DIM, Wlds[cur ^ 1], tid);

        long g0 = (long)nbr[r0 * KTAP + tap];
        long g1 = (long)nbr[r1 * KTAP + tap];
        const v8bf* fr0 = (const v8bf*)(Fb + g0 * C_DIM);
        const v8bf* fr1 = (const v8bf*)(Fb + g1 * C_DIM);
        const bf16* wt  = Wlds[cur];

        #pragma unroll
        for (int kk = 0; kk < C_DIM; kk += 32) {
            const int ci = (kk >> 3) + hl;
            v16bf a0 = cat8(fr0[ci], fr0[ci + 2]);
            v16bf a1 = cat8(fr1[ci], fr1[ci + 2]);
            #pragma unroll
            for (int nt = 0; nt < 4; ++nt) {
                const v8bf* bp = (const v8bf*)&wt[(nt * 16 + lr) * C_DIM + kk + 8 * hl];
                v16bf b = cat8(bp[0], bp[2]);
                acc[0][nt] = wmma_bf16(a0, b, acc[0][nt]);
                acc[1][nt] = wmma_bf16(a1, b, acc[1][nt]);
            }
        }
        stage_wait();   // next buffer complete + all waves past this tap
    }
    #pragma unroll
    for (int s = 0; s < 2; ++s) {
        #pragma unroll
        for (int nt = 0; nt < 4; ++nt) {
            #pragma unroll
            for (int r = 0; r < 8; ++r) {
                long row = m0 + s * 16 + r + 8 * hl;
                if (row < nrows) Y[row * C_DIM + nt * 16 + lr] = acc[s][nt][r];
            }
        }
    }
}

// ---------------------------------------------------------------------------
// BatchNorm (training-mode) statistics: per-channel sum / sumsq via atomics
// ---------------------------------------------------------------------------
__global__ void colstats_kernel(const float* __restrict__ X, float* __restrict__ stats, long nrows) {
    int c  = threadIdx.x & 63;
    int rg = threadIdx.x >> 6;
    long r = (long)blockIdx.x * 4 + rg;
    long stride = (long)gridDim.x * 4;
    float s = 0.0f, q = 0.0f;
    for (; r < nrows; r += stride) {
        float v = X[r * C_DIM + c];
        s += v; q += v * v;
    }
    atomicAdd(&stats[c], s);
    atomicAdd(&stats[C_DIM + c], q);
}

// mode 0: relu(bn(x));  mode 1: relu(bn(x)) + res;  mode 2: relu(bn(x) + res)
__global__ void bn_elem_kernel(const float* __restrict__ X, const float* __restrict__ stats,
                               const float* __restrict__ g, const float* __restrict__ b,
                               const float* __restrict__ res, float* __restrict__ Y,
                               long count, int mode, float invN) {
    long i = (long)blockIdx.x * blockDim.x + threadIdx.x;
    if (i >= count) return;
    int c = (int)(i & 63);
    float mu  = stats[c] * invN;
    float var = fmaxf(stats[C_DIM + c] * invN - mu * mu, 0.0f);
    float xn  = (X[i] - mu) * rsqrtf(var + 1e-3f) * g[c] + b[c];
    float o;
    if (mode == 0)      o = fmaxf(xn, 0.0f);
    else if (mode == 1) o = fmaxf(xn, 0.0f) + res[i];
    else                o = fmaxf(xn + res[i], 0.0f);
    Y[i] = o;
}

// ---------------------------------------------------------------------------
// Segment (cluster) ops
// ---------------------------------------------------------------------------
__global__ void seg_cnt_kernel(const int* __restrict__ cl, float* __restrict__ cnt, long nrows) {
    long n = (long)blockIdx.x * blockDim.x + threadIdx.x;
    if (n < nrows) atomicAdd(&cnt[cl[n]], 1.0f);
}

__global__ void seg_sum_kernel(const float* __restrict__ X, const int* __restrict__ cl,
                               float* __restrict__ acc, long nrows) {
    long i = (long)blockIdx.x * blockDim.x + threadIdx.x;
    if (i >= nrows * C_DIM) return;
    long n = i >> 6; int c = (int)(i & 63);
    atomicAdd(&acc[(long)cl[n] * C_DIM + c], X[i]);
}

__global__ void sub_mean_kernel(float* __restrict__ X, const int* __restrict__ cl,
                                const float* __restrict__ acc, const float* __restrict__ cnt,
                                long nrows) {
    long i = (long)blockIdx.x * blockDim.x + threadIdx.x;
    if (i >= nrows * C_DIM) return;
    long n = i >> 6; int c = (int)(i & 63);
    long s = cl[n];
    X[i] -= acc[s * C_DIM + c] / fmaxf(cnt[s], 1.0f);
}

__global__ void gmax_kernel(const float* __restrict__ X, unsigned* __restrict__ gmax, long count) {
    __shared__ unsigned sm[256];
    long i = (long)blockIdx.x * blockDim.x + threadIdx.x;
    long stride = (long)gridDim.x * blockDim.x;
    unsigned loc = 0u;
    for (; i < count; i += stride) {
        unsigned e = fenc(X[i]);
        loc = (loc > e) ? loc : e;
    }
    sm[threadIdx.x] = loc;
    __syncthreads();
    for (int s = 128; s > 0; s >>= 1) {
        if ((int)threadIdx.x < s) {
            unsigned o = sm[threadIdx.x + s];
            if (o > sm[threadIdx.x]) sm[threadIdx.x] = o;
        }
        __syncthreads();
    }
    if (threadIdx.x == 0) atomicMax(gmax, sm[0]);
}

__global__ void exp_kernel(float* __restrict__ X, const unsigned* __restrict__ gmax, long count) {
    long i = (long)blockIdx.x * blockDim.x + threadIdx.x;
    if (i >= count) return;
    float m = fdec(*gmax);
    X[i] = __expf(X[i] - m);
}

__global__ void div_seg_kernel(float* __restrict__ X, const int* __restrict__ cl,
                               const float* __restrict__ acc, long nrows) {
    long i = (long)blockIdx.x * blockDim.x + threadIdx.x;
    if (i >= nrows * C_DIM) return;
    long n = i >> 6; int c = (int)(i & 63);
    X[i] = X[i] / (acc[(long)cl[n] * C_DIM + c] + 1e-6f);
}

__global__ void gather_seg_kernel(const float* __restrict__ acc, const int* __restrict__ cl,
                                  float* __restrict__ Y, long nrows) {
    long i = (long)blockIdx.x * blockDim.x + threadIdx.x;
    if (i >= nrows * C_DIM) return;
    long n = i >> 6; int c = (int)(i & 63);
    Y[i] = acc[(long)cl[n] * C_DIM + c];
}

__global__ void mul_kernel(float* __restrict__ Y, const float* __restrict__ A,
                           const float* __restrict__ B, long count) {
    long i = (long)blockIdx.x * blockDim.x + threadIdx.x;
    if (i < count) Y[i] = A[i] * B[i];
}

// adp = softmax(feat @ W_adp) over 3 scales; mix = sum_s adp[s] * pf_s
__global__ void adp_mix_kernel(const float* __restrict__ feat, const float* __restrict__ Wadp,
                               const float* __restrict__ pf0, const float* __restrict__ pf1,
                               const float* __restrict__ pf2, float* __restrict__ mix, long nrows) {
    long n = (long)blockIdx.x * blockDim.x + threadIdx.x;
    if (n >= nrows) return;
    const float* fr = feat + n * C_DIM;
    float s0 = 0.f, s1 = 0.f, s2 = 0.f;
    #pragma unroll
    for (int c = 0; c < C_DIM; ++c) {
        float v = fr[c];
        s0 += v * Wadp[c * 3 + 0];
        s1 += v * Wadp[c * 3 + 1];
        s2 += v * Wadp[c * 3 + 2];
    }
    float mx = fmaxf(s0, fmaxf(s1, s2));
    float e0 = __expf(s0 - mx), e1 = __expf(s1 - mx), e2 = __expf(s2 - mx);
    float inv = 1.0f / (e0 + e1 + e2);
    e0 *= inv; e1 *= inv; e2 *= inv;
    #pragma unroll 4
    for (int c = 0; c < C_DIM; ++c)
        mix[n * C_DIM + c] = e0 * pf0[n * C_DIM + c] + e1 * pf1[n * C_DIM + c] + e2 * pf2[n * C_DIM + c];
}

// ---------------------------------------------------------------------------
// Host orchestration
// ---------------------------------------------------------------------------
extern "C" void kernel_launch(void* const* d_in, const int* in_sizes, int n_in,
                              void* d_out, int out_size, void* d_ws, size_t ws_size,
                              hipStream_t stream) {
    const float* feat   = (const float*)d_in[0];
    const int*   cl[3]  = {(const int*)d_in[1], (const int*)d_in[2], (const int*)d_in[3]};
    const int*   nbr    = (const int*)d_in[4];
    const float* W_lw   = (const float*)d_in[5];
    const float* g_lw   = (const float*)d_in[6];
    const float* b_lw   = (const float*)d_in[7];
    const float* W_w    = (const float*)d_in[8];
    const float* W_proj = (const float*)d_in[9];
    const float* g_proj = (const float*)d_in[10];
    const float* b_proj = (const float*)d_in[11];
    const float* W_adp  = (const float*)d_in[12];
    const float* W_fuse = (const float*)d_in[13];
    const float* g_fuse = (const float*)d_in[14];
    const float* b_fuse = (const float*)d_in[15];
    const float* W_c1   = (const float*)d_in[16];
    const float* g_c1   = (const float*)d_in[17];
    const float* b_c1   = (const float*)d_in[18];
    const float* W_c2   = (const float*)d_in[19];
    const float* g_c2   = (const float*)d_in[20];
    const float* b_c2   = (const float*)d_in[21];

    const long N = (long)in_sizes[0] / C_DIM;
    const long NCLs[3] = {4096, 32768, 131072};
    const long NCLMAX = 131072;

    // ---- workspace carve-up (256B aligned) ----
    char* base = (char*)d_ws;
    size_t off = 0;
    auto carve = [&](size_t bytes) -> char* {
        char* p = base + off;
        off = (off + bytes + 255) & ~(size_t)255;
        return p;
    };
    bf16* wb_lw   = (bf16*)carve(3 * 4096 * sizeof(bf16));     // transposed panels
    bf16* wb_w    = (bf16*)carve(3 * 4096 * sizeof(bf16));
    bf16* wb_proj = (bf16*)carve(4 * 4096 * sizeof(bf16));
    bf16* wb_fuse = (bf16*)carve(8192 * sizeof(bf16));
    bf16* wb_c1   = (bf16*)carve((size_t)KTAP * 4096 * sizeof(bf16));
    bf16* wb_c2   = (bf16*)carve((size_t)KTAP * 4096 * sizeof(bf16));
    float*    stats  = (float*)carve(128 * sizeof(float));
    unsigned* gmax   = (unsigned*)carve(sizeof(unsigned));
    float*    cnt    = (float*)carve((size_t)NCLMAX * sizeof(float));
    float*    segacc = (float*)carve((size_t)NCLMAX * C_DIM * sizeof(float));
    size_t nc = (size_t)N * C_DIM * sizeof(float);
    float* B_pw    = (float*)carve(nc);
    float* B_tmp   = (float*)carve(nc);
    float* B_pf0   = (float*)carve(nc);
    float* B_pf1   = (float*)carve(nc);
    float* B_pf2   = (float*)carve(nc);
    float* B_fused = (float*)carve(nc);
    bf16*  B_fb    = (bf16*)carve((size_t)N * C_DIM * sizeof(bf16)); // bf16 feature cache
    float* B_pf[3] = {B_pf0, B_pf1, B_pf2};

    const long EW = N * C_DIM;
    const int  ew_blk = (int)((EW + 255) / 256);
    const int  n_blk  = (int)((N + 255) / 256);
    const int  g_blk  = (int)((N + 127) / 128);
    const int  c_blk  = (int)((N + 255) / 256);
    const float invN  = 1.0f / (float)N;

    auto cvtT = [&](const float* s, bf16* d, int kdim, long cnt_) {
        cvt_bf16_T_kernel<<<(int)((cnt_ + 255) / 256), 256, 0, stream>>>(s, d, kdim, cnt_);
    };
    auto bn_apply = [&](const float* X, const float* gg, const float* bb,
                        const float* res, float* Y, int mode) {
        (void)hipMemsetAsync(stats, 0, 128 * sizeof(float), stream);
        colstats_kernel<<<1024, 256, 0, stream>>>(X, stats, N);
        bn_elem_kernel<<<ew_blk, 256, 0, stream>>>(X, stats, gg, bb, res, Y, EW, mode, invN);
    };

    // ---- weights -> transposed bf16 panels ----
    cvtT(W_lw,   wb_lw,   64, 3 * 4096);
    cvtT(W_w,    wb_w,    64, 3 * 4096);
    cvtT(W_proj, wb_proj, 64, 4 * 4096);
    cvtT(W_fuse, wb_fuse, 128, 8192);
    cvtT(W_c1,   wb_c1,   64, (long)KTAP * 4096);
    cvtT(W_c2,   wb_c2,   64, (long)KTAP * 4096);

    // ---- three attention scales ----
    for (int i = 0; i < 3; ++i) {
        long ncl = NCLs[i];
        // pw = relu(bn(feat @ W_lw[i]))
        gemm_wmma_kernel<64><<<g_blk, 256, 0, stream>>>(feat, feat, wb_lw + (long)i * 4096, B_pw, N);
        bn_apply(B_pw, g_lw + i * 64, b_lw + i * 64, B_pw, B_pw, 0);
        // pw -= cluster mean
        (void)hipMemsetAsync(cnt, 0, (size_t)ncl * sizeof(float), stream);
        (void)hipMemsetAsync(segacc, 0, (size_t)ncl * C_DIM * sizeof(float), stream);
        seg_cnt_kernel<<<n_blk, 256, 0, stream>>>(cl[i], cnt, N);
        seg_sum_kernel<<<ew_blk, 256, 0, stream>>>(B_pw, cl[i], segacc, N);
        sub_mean_kernel<<<ew_blk, 256, 0, stream>>>(B_pw, cl[i], segacc, cnt, N);
        // pw = scatter-softmax(pw @ W_w[i]) per cluster
        gemm_wmma_kernel<64><<<g_blk, 256, 0, stream>>>(B_pw, B_pw, wb_w + (long)i * 4096, B_tmp, N);
        (void)hipMemsetAsync(gmax, 0, sizeof(unsigned), stream);
        gmax_kernel<<<1024, 256, 0, stream>>>(B_tmp, gmax, EW);
        exp_kernel<<<ew_blk, 256, 0, stream>>>(B_tmp, gmax, EW);
        (void)hipMemsetAsync(segacc, 0, (size_t)ncl * C_DIM * sizeof(float), stream);
        seg_sum_kernel<<<ew_blk, 256, 0, stream>>>(B_tmp, cl[i], segacc, N);
        div_seg_kernel<<<ew_blk, 256, 0, stream>>>(B_tmp, cl[i], segacc, N);
        // pfeat = gather(segsum(relu(bn(feat @ W_proj[i])) * pw))
        gemm_wmma_kernel<64><<<g_blk, 256, 0, stream>>>(feat, feat, wb_proj + (long)i * 4096, B_pw, N);
        bn_apply(B_pw, g_proj + i * 64, b_proj + i * 64, B_pw, B_pw, 0);
        mul_kernel<<<ew_blk, 256, 0, stream>>>(B_pw, B_pw, B_tmp, EW);
        (void)hipMemsetAsync(segacc, 0, (size_t)ncl * C_DIM * sizeof(float), stream);
        seg_sum_kernel<<<ew_blk, 256, 0, stream>>>(B_pw, cl[i], segacc, N);
        gather_seg_kernel<<<ew_blk, 256, 0, stream>>>(segacc, cl[i], B_pf[i], N);
    }

    // ---- adaptive mix + fuse ----
    adp_mix_kernel<<<n_blk, 256, 0, stream>>>(feat, W_adp, B_pf0, B_pf1, B_pf2, B_tmp, N); // mix
    gemm_wmma_kernel<64><<<g_blk, 256, 0, stream>>>(feat, feat, wb_proj + 3 * 4096, B_pw, N);
    bn_apply(B_pw, g_proj + 192, b_proj + 192, B_pw, B_pw, 0);                              // f_last
    gemm_wmma_kernel<128><<<g_blk, 256, 0, stream>>>(B_pw, B_tmp, wb_fuse, B_pf0, N);       // concat GEMM
    bn_apply(B_pf0, g_fuse, b_fuse, feat, B_fused, 1);                                      // relu(bn)+feat

    // ---- voxel block: conv -> bn -> relu -> conv -> bn; out = relu(h + res) ----
    cvt_bf16_kernel<<<ew_blk, 256, 0, stream>>>(B_fused, B_fb, EW);
    subm_conv_wmma_kernel<<<c_blk, 256, 0, stream>>>(B_fb, nbr, wb_c1, B_pw, N);
    bn_apply(B_pw, g_c1, b_c1, B_pw, B_pw, 0);
    cvt_bf16_kernel<<<ew_blk, 256, 0, stream>>>(B_pw, B_fb, EW);
    subm_conv_wmma_kernel<<<c_blk, 256, 0, stream>>>(B_fb, nbr, wb_c2, B_tmp, N);
    bn_apply(B_tmp, g_c2, b_c2, B_fused, (float*)d_out, 2);
}
